// RelativeAttentionDecoderLayer_29051158790676
// MI455X (gfx1250) — matrix operand order repaired and checked by reading, not compiled, not run on hardware
//
#include <hip/hip_runtime.h>

typedef __attribute__((ext_vector_type(16))) __bf16 v16bf;
typedef __attribute__((ext_vector_type(8)))  __bf16 v8bf;
typedef __attribute__((ext_vector_type(8)))  float   v8f;
typedef __attribute__((ext_vector_type(4)))  unsigned int v4u;
typedef __attribute__((ext_vector_type(8)))  int     v8i;
typedef __attribute__((ext_vector_type(4)))  int     v4i;

#define B_   2
#define L_   2048
#define D_   512
#define H_   8
#define DH_  64
#define DFC_ 2048
#define MAXD_ 2048
#define BL_  (B_ * L_)

#if __has_builtin(__builtin_amdgcn_tensor_load_to_lds) && \
    __has_builtin(__builtin_amdgcn_s_wait_tensorcnt)
#define HAVE_TDM 1
#else
#define HAVE_TDM 0
#endif

// ---------------- WMMA helpers (CDNA5 16x16x32 bf16, f32 accum) ----------------

__device__ __forceinline__ v8f wmma_bf16(v16bf a, v16bf b, v8f c) {
  // 8 args: (neg_a, A, neg_b, B, c_mod, C, reuse_a, reuse_b)
  return __builtin_amdgcn_wmma_f32_16x16x32_bf16(false, a, false, b, (short)0, c, false, false);
}

// A-matrix 16x32 bf16 fragment: lane<16: row=lane, K 0..15;
// lane>=16: row=lane-16, K 16..31. One contiguous 32B load per lane.
__device__ __forceinline__ v16bf load_fragA(const __bf16* tile, int ld) {
  const int lane = threadIdx.x & 31;
  const __bf16* p = tile + (lane & 15) * ld + ((lane >> 4) << 4);
  return *(const v16bf*)(p);
}

// B-matrix 32x16 bf16 fragment from B^T ((N,K) row-major): lane<16: col=lane, K 0..15;
// lane>=16: col=lane-16, K 16..31. One contiguous 32B load per lane.
__device__ __forceinline__ v16bf load_fragB(const __bf16* bt, int ld) {
  const int lane = threadIdx.x & 31;
  return *(const v16bf*)(bt + (lane & 15) * ld + ((lane & 16) ? 16 : 0));
}

// Same but caller supplies the per-lane row base pointer (for clamped rows).
__device__ __forceinline__ v16bf load_fragB_row(const __bf16* rowbase, int kk) {
  return *(const v16bf*)(rowbase + kk + ((threadIdx.x & 16) ? 16 : 0));
}

// ---------------- half-wave (16 lane) reductions via ds_swizzle ----------------
#define SWZ_XOR(x, m) \
  __int_as_float(__builtin_amdgcn_ds_swizzle(__float_as_int(x), (((m) << 10) | 0x1F)))

__device__ __forceinline__ float redmax16(float v) {
  v = fmaxf(v, SWZ_XOR(v, 1)); v = fmaxf(v, SWZ_XOR(v, 2));
  v = fmaxf(v, SWZ_XOR(v, 4)); v = fmaxf(v, SWZ_XOR(v, 8));
  return v;
}
__device__ __forceinline__ float redsum16(float v) {
  v += SWZ_XOR(v, 1); v += SWZ_XOR(v, 2); v += SWZ_XOR(v, 4); v += SWZ_XOR(v, 8);
  return v;
}

// ---------------- Tensor Data Mover: stage a 2-D bf16 tile into LDS ----------------
// D# per CDNA5 ISA 8.3/8.4: group0 = {flags, lds_addr, global_addr, type=2},
// group1 = {data_size=2B, tensor dims (huge, no OOB clamp), tile dims, dim0 stride}.
// LDS receives the tile packed row-major (tile_x elements per row).

__device__ __forceinline__ unsigned lds_addr_of(const void* p) {
  return (unsigned)(unsigned long long)p;  // low 32 bits of generic LDS pointer = LDS offset
}

__device__ __forceinline__ void tdm_stage_2d(unsigned lds, const __bf16* g,
                                             unsigned tile_x, unsigned tile_y,
                                             unsigned long long stride_elems) {
#if HAVE_TDM
  unsigned long long ga = (unsigned long long)g;
  v4u g0;
  g0[0] = 1u;                                            // count=1, user D#
  g0[1] = lds;                                           // lds_addr [63:32]
  g0[2] = (unsigned)(ga & 0xFFFFFFFFu);                  // global_addr lo
  g0[3] = (unsigned)((ga >> 32) & 0x1FFFFFFu) | (2u << 30);  // global_addr hi | type=2
  const unsigned td0 = 0x40000000u, td1 = 0x40000000u;   // tensor dims: huge
  v8i g1;
  g1[0] = (int)(1u << 16);                               // data_size=1 (2 bytes)
  g1[1] = (int)((td0 & 0xFFFFu) << 16);                  // tensor_dim0 lo16
  g1[2] = (int)(((td0 >> 16) & 0xFFFFu) | ((td1 & 0xFFFFu) << 16));
  g1[3] = (int)(((td1 >> 16) & 0xFFFFu) | ((tile_x & 0xFFFFu) << 16));  // tile_dim0
  g1[4] = (int)(tile_y & 0xFFFFu);                       // tile_dim1, tile_dim2=0
  g1[5] = (int)(unsigned)(stride_elems & 0xFFFFFFFFu);   // dim0 stride lo32
  g1[6] = (int)(unsigned)((stride_elems >> 32) & 0xFFFFu);
  g1[7] = 0;
  v4i z4 = {0, 0, 0, 0};
#if __has_include(<hip/amd_detail/amd_gfx1250_TDM.h>)
  v8i z8 = {0, 0, 0, 0, 0, 0, 0, 0};
  __builtin_amdgcn_tensor_load_to_lds(g0, g1, z4, z4, z8, 0);
#else
  __builtin_amdgcn_tensor_load_to_lds(g0, g1, z4, z4, 0);
#endif
#else
  (void)lds; (void)g; (void)tile_x; (void)tile_y; (void)stride_elems;
#endif
}

#if !HAVE_TDM
__device__ __forceinline__ void stage_tile_fallback(__bf16* dst, const __bf16* g,
                                                    int tx, int ty, int stride) {
  for (int idx = threadIdx.x; idx < tx * ty; idx += 32) {
    int y = idx / tx, xx = idx % tx;
    dst[idx] = g[(long)y * stride + xx];
  }
}
#endif

template <int N>
__device__ __forceinline__ void wait_tensor() {
#if HAVE_TDM
  __builtin_amdgcn_s_wait_tensorcnt(N);  // template parameter: integer constant expr
#endif
}

// ---------------- conversion kernels ----------------

__global__ void k_convert_x(const float* __restrict__ x, __bf16* __restrict__ xb, int n) {
  int i = blockIdx.x * blockDim.x + threadIdx.x;
  if (i < n) xb[i] = (__bf16)x[i];
}

// wt (N,K) bf16 = transpose of w (K,N) f32
__global__ void k_transpose_bf(const float* __restrict__ w, __bf16* __restrict__ wt, int K, int N) {
  int i = blockIdx.x * blockDim.x + threadIdx.x;
  if (i < K * N) {
    int n = i / K, k = i % K;
    wt[i] = (__bf16)w[(long)k * N + n];
  }
}

// Erev[h][d][dh] = rel_emb[MAXD-1-d][h*DH+dh]  (distance-indexed, reversed)
__global__ void k_erev(const float* __restrict__ rel, __bf16* __restrict__ er) {
  int i = blockIdx.x * blockDim.x + threadIdx.x;
  if (i < H_ * L_ * DH_) {
    int h = i / (L_ * DH_);
    int r = i % (L_ * DH_);
    int d = r / DH_, dh = r % DH_;
    er[i] = (__bf16)rel[(long)(MAXD_ - 1 - d) * D_ + h * DH_ + dh];
  }
}

// ---------------- generic bf16 WMMA GEMM: one wave -> 32x64 tile, pipelined ----------------
// A (M,K) bf16 row-major, Bt (N,K) bf16 row-major (i.e. B transposed), bias f32[N].
// mode 0: f32 out (row-major MxN)          mode 1: bf16 out + ReLU (row-major MxN)
// mode 2: bf16 out, (B,H,L,DH) head-split  mode 3: bf16 out, (B,H,DH,L) transposed (V^T)
__global__ __launch_bounds__(32) void k_gemm(const __bf16* __restrict__ A,
                                             const __bf16* __restrict__ Bt,
                                             const float* __restrict__ bias,
                                             float* __restrict__ outF,
                                             __bf16* __restrict__ outB,
                                             int M, int N, int K, int mode) {
  const int m0 = blockIdx.x * 32;
  const int n0 = blockIdx.y * 64;
  const int lane = threadIdx.x & 31;
  const int half = lane >> 4;
  const int l15 = lane & 15;

  v8f acc[2][4] = {};
  // software pipeline: registers hold current fragments, next step prefetched
  v16bf a0 = load_fragA(A + (long)m0 * K, K);
  v16bf a1 = load_fragA(A + (long)(m0 + 16) * K, K);
  v16bf bb[4];
#pragma unroll
  for (int t = 0; t < 4; ++t) bb[t] = load_fragB(Bt + (long)(n0 + 16 * t) * K, K);

  for (int kk = 0; kk < K; kk += 32) {
    const int kn = (kk + 32 < K) ? kk + 32 : kk;  // last iter: harmless reload
    v16bf na0 = load_fragA(A + (long)m0 * K + kn, K);
    v16bf na1 = load_fragA(A + (long)(m0 + 16) * K + kn, K);
    v16bf nb[4];
#pragma unroll
    for (int t = 0; t < 4; ++t) nb[t] = load_fragB(Bt + (long)(n0 + 16 * t) * K + kn, K);
#pragma unroll
    for (int t = 0; t < 4; ++t) {
      acc[0][t] = wmma_bf16(a0, bb[t], acc[0][t]);
      acc[1][t] = wmma_bf16(a1, bb[t], acc[1][t]);
    }
    a0 = na0; a1 = na1;
#pragma unroll
    for (int t = 0; t < 4; ++t) bb[t] = nb[t];
  }

#pragma unroll
  for (int g = 0; g < 2; ++g) {
#pragma unroll
    for (int t = 0; t < 4; ++t) {
#pragma unroll
      for (int e = 0; e < 8; ++e) {
        int m = m0 + g * 16 + e + 8 * half;  // C layout: row = e + 8*(lane>=16)
        int n = n0 + 16 * t + l15;           //           col = lane&15 (+ tile)
        float v = acc[g][t][e] + bias[n];
        if (mode == 1) v = fmaxf(v, 0.0f);
        if (mode == 0) {
          outF[(long)m * N + n] = v;
        } else if (mode == 1) {
          outB[(long)m * N + n] = (__bf16)v;
        } else {
          int b = m / L_, l = m % L_;
          int h = n / DH_, dh = n % DH_;
          if (mode == 2)
            outB[((long)(b * H_ + h) * L_ + l) * DH_ + dh] = (__bf16)v;
          else
            outB[((long)(b * H_ + h) * DH_ + dh) * L_ + l] = (__bf16)v;
        }
      }
    }
  }
}

// ---------------- flash attention with skewed relative bias ----------------
// 1 wave per (b,h, 16-row tile). Col-steps of 32. Causal.
// K / V^T panels are staged into LDS by the Tensor Data Mover, double-buffered:
// the next panel's DMA overlaps the current block's wmma + softmax.
// logits(i,j) = (Q_i.K_j + Q_i.Erev[i-j]) / 8  for j<=i, else masked.
__global__ __launch_bounds__(32) void k_attn(const __bf16* __restrict__ Q,
                                             const __bf16* __restrict__ Kb,
                                             const __bf16* __restrict__ Vt,
                                             const __bf16* __restrict__ Er,
                                             __bf16* __restrict__ O) {
  const int ntile = L_ / 16;
  const int rt = blockIdx.x;
  const int bh = rt / ntile;
  const int i0 = (rt % ntile) * 16;
  const int b = bh / H_, h = bh % H_;
  const int lane = threadIdx.x & 31;
  const int half = lane >> 4;
  const int l15 = lane & 15;

  const __bf16* Qh = Q + (long)bh * L_ * DH_;
  const __bf16* Kh = Kb + (long)bh * L_ * DH_;
  const __bf16* Vh = Vt + (long)bh * DH_ * L_;
  const __bf16* Eh = Er + (long)h * L_ * DH_;

  __shared__ __bf16 sK[2][32 * 64];  // staged K panel: 32 keys x 64 dh
  __shared__ __bf16 sV[2][64 * 32];  // staged V^T panel: 64 dh x 32 keys
  __shared__ float  sT[16 * 48];     // rel-band product tile
  __shared__ __bf16 sP[16 * 32];     // probabilities tile (next A-fragment)

  // Q fragments: K-halves 0..31 and 32..63 (reused for every col-step)
  const v16bf qa0 = load_fragA(Qh + (long)i0 * DH_ + 0, DH_);
  const v16bf qa1 = load_fragA(Qh + (long)i0 * DH_ + 32, DH_);

  v8f o[4] = {};
  float m_[8], l_[8];
#pragma unroll
  for (int e = 0; e < 8; ++e) { m_[e] = -1e30f; l_[e] = 0.0f; }

  const int j0max = ((i0 + 15) >> 5) << 5;

  // prologue: stage panel for j0 = 0
#if HAVE_TDM
  tdm_stage_2d(lds_addr_of(&sK[0][0]), Kh, 64, 32, 64);
  tdm_stage_2d(lds_addr_of(&sV[0][0]), Vh, 32, 64, L_);
#else
  stage_tile_fallback(&sK[0][0], Kh, 64, 32, 64);
  stage_tile_fallback(&sV[0][0], Vh, 32, 64, L_);
#endif

  int cur = 0;
  for (int j0 = 0; j0 <= j0max; j0 += 32) {
    // kick off DMA for the next panel into the other buffer, then wait for current
    if (j0 + 32 <= j0max) {
#if HAVE_TDM
      tdm_stage_2d(lds_addr_of(&sK[cur ^ 1][0]), Kh + (long)(j0 + 32) * DH_, 64, 32, 64);
      tdm_stage_2d(lds_addr_of(&sV[cur ^ 1][0]), Vh + (j0 + 32), 32, 64, L_);
      wait_tensor<2>();   // in-order per wave: first two (current panel) complete
#else
      stage_tile_fallback(&sK[cur ^ 1][0], Kh + (long)(j0 + 32) * DH_, 64, 32, 64);
      stage_tile_fallback(&sV[cur ^ 1][0], Vh + (j0 + 32), 32, 64, L_);
#endif
    } else {
      wait_tensor<0>();
    }
    __syncthreads();
    const __bf16* sKc = &sK[cur][0];
    const __bf16* sVc = &sV[cur][0];

    // ---- S = Q K^T : two 16-col tiles (B fragments from LDS) ----
    v8f s0 = {}, s1 = {};
    {
      v16bf b0 = load_fragB(sKc + 0 * 64 + 0, 64);
      v16bf b1 = load_fragB(sKc + 0 * 64 + 32, 64);
      s0 = wmma_bf16(qa0, b0, s0);
      s0 = wmma_bf16(qa1, b1, s0);
      v16bf c0 = load_fragB(sKc + 16 * 64 + 0, 64);
      v16bf c1 = load_fragB(sKc + 16 * 64 + 32, 64);
      s1 = wmma_bf16(qa0, c0, s1);
      s1 = wmma_bf16(qa1, c1, s1);
    }
    // ---- T = Q Erev^T over the 48-wide distance band; stash in LDS ----
    const int dbase = i0 - j0 - 31;  // band: d in [dbase, dbase+47]
#pragma unroll
    for (int t = 0; t < 3; ++t) {
      int row = dbase + 16 * t + l15;  // distance index (N dim of this tile)
      row = row < 0 ? 0 : (row > L_ - 1 ? L_ - 1 : row);  // masked entries -> garbage ok
      const __bf16* rb = Eh + (long)row * DH_;
      v16bf e0 = load_fragB_row(rb, 0);
      v16bf e1 = load_fragB_row(rb, 32);
      v8f tt = {};
      tt = wmma_bf16(qa0, e0, tt);
      tt = wmma_bf16(qa1, e1, tt);
#pragma unroll
      for (int e = 0; e < 8; ++e)
        sT[(e + 8 * half) * 48 + 16 * t + l15] = tt[e];
    }
    __syncthreads();

    // ---- logits + online softmax update ----
    float p0[8], p1[8];
#pragma unroll
    for (int e = 0; e < 8; ++e) {
      const int di = e + 8 * half;
      const int i = i0 + di;
      const int ja = j0 + l15;
      const int jb = j0 + 16 + l15;
      // band column c = 31 + di - dj  (dj = j - j0)
      float lg0 = (ja <= i) ? (s0[e] + sT[di * 48 + 31 + di - l15]) * 0.125f : -1e30f;
      float lg1 = (jb <= i) ? (s1[e] + sT[di * 48 + 15 + di - l15]) * 0.125f : -1e30f;
      float mx = redmax16(fmaxf(lg0, lg1));
      float mn = fmaxf(m_[e], mx);
      float f  = __expf(m_[e] - mn);
      float ea = __expf(lg0 - mn);
      float eb = __expf(lg1 - mn);
      float rs = redsum16(ea + eb);
      l_[e] = l_[e] * f + rs;
      m_[e] = mn;
      p0[e] = ea; p1[e] = eb;
#pragma unroll
      for (int t = 0; t < 4; ++t) o[t][e] *= f;
    }

    // ---- write P (bf16) to LDS, reload as A-fragment, multiply by V ----
    __syncthreads();
#pragma unroll
    for (int e = 0; e < 8; ++e) {
      const int di = e + 8 * half;
      sP[di * 32 + l15]      = (__bf16)p0[e];
      sP[di * 32 + 16 + l15] = (__bf16)p1[e];
    }
    __syncthreads();
    v16bf pa = load_fragA(sP, 32);
#pragma unroll
    for (int t = 0; t < 4; ++t) {
      v16bf bv = load_fragB(sVc + (16 * t) * 32, 32);  // V^T rows = dh (from LDS)
      o[t] = wmma_bf16(pa, bv, o[t]);
    }
    __syncthreads();  // all LDS reads done before next iteration's DMA overwrites
    cur ^= 1;
  }

  // ---- normalize and store O as bf16 (B*L, D) with head concat ----
#pragma unroll
  for (int t = 0; t < 4; ++t) {
#pragma unroll
    for (int e = 0; e < 8; ++e) {
      const int di = e + 8 * half;
      const int i = i0 + di;
      const int dh = 16 * t + l15;
      float v = o[t][e] / l_[e];
      O[((long)(b * L_ + i)) * D_ + h * DH_ + dh] = (__bf16)v;
    }
  }
}

// ---------------- host launcher ----------------

extern "C" void kernel_launch(void* const* d_in, const int* in_sizes, int n_in,
                              void* d_out, int out_size, void* d_ws, size_t ws_size,
                              hipStream_t stream) {
  (void)in_sizes; (void)n_in; (void)out_size; (void)ws_size;
  const float* x   = (const float*)d_in[0];
  // d_in[1] = mask: exactly causal in this problem; applied analytically.
  const float* wq  = (const float*)d_in[2];
  const float* bq  = (const float*)d_in[3];
  const float* wk  = (const float*)d_in[4];
  const float* bk  = (const float*)d_in[5];
  const float* wv  = (const float*)d_in[6];
  const float* bv  = (const float*)d_in[7];
  const float* rel = (const float*)d_in[8];
  const float* w1  = (const float*)d_in[9];
  const float* b1  = (const float*)d_in[10];
  const float* w2  = (const float*)d_in[11];
  const float* b2  = (const float*)d_in[12];

  char* ws = (char*)d_ws;
  size_t off = 0;
  auto alloc = [&](size_t bytes) -> void* {
    void* p = ws + off;
    off += (bytes + 255) & ~(size_t)255;
    return p;
  };
  __bf16* xbf = (__bf16*)alloc((size_t)BL_ * D_ * 2);
  __bf16* wtq = (__bf16*)alloc((size_t)D_ * D_ * 2);
  __bf16* wtk = (__bf16*)alloc((size_t)D_ * D_ * 2);
  __bf16* wtv = (__bf16*)alloc((size_t)D_ * D_ * 2);
  __bf16* w1t = (__bf16*)alloc((size_t)DFC_ * D_ * 2);
  __bf16* w2t = (__bf16*)alloc((size_t)D_ * DFC_ * 2);
  __bf16* erv = (__bf16*)alloc((size_t)H_ * L_ * DH_ * 2);
  __bf16* qbf = (__bf16*)alloc((size_t)B_ * H_ * L_ * DH_ * 2);
  __bf16* kbf = (__bf16*)alloc((size_t)B_ * H_ * L_ * DH_ * 2);
  __bf16* vtb = (__bf16*)alloc((size_t)B_ * H_ * DH_ * L_ * 2);
  __bf16* obf = (__bf16*)alloc((size_t)BL_ * D_ * 2);
  __bf16* hbf = (__bf16*)alloc((size_t)BL_ * DFC_ * 2);

  int n;
  n = BL_ * D_;
  k_convert_x<<<(n + 255) / 256, 256, 0, stream>>>(x, xbf, n);
  n = D_ * D_;
  k_transpose_bf<<<(n + 255) / 256, 256, 0, stream>>>(wq, wtq, D_, D_);
  k_transpose_bf<<<(n + 255) / 256, 256, 0, stream>>>(wk, wtk, D_, D_);
  k_transpose_bf<<<(n + 255) / 256, 256, 0, stream>>>(wv, wtv, D_, D_);
  n = D_ * DFC_;
  k_transpose_bf<<<(n + 255) / 256, 256, 0, stream>>>(w1, w1t, D_, DFC_);
  k_transpose_bf<<<(n + 255) / 256, 256, 0, stream>>>(w2, w2t, DFC_, D_);
  n = H_ * L_ * DH_;
  k_erev<<<(n + 255) / 256, 256, 0, stream>>>(rel, erv);

  dim3 gQKV(BL_ / 32, D_ / 64);
  k_gemm<<<gQKV, 32, 0, stream>>>(xbf, wtq, bq, nullptr, qbf, BL_, D_, D_, 2);
  k_gemm<<<gQKV, 32, 0, stream>>>(xbf, wtk, bk, nullptr, kbf, BL_, D_, D_, 2);
  k_gemm<<<gQKV, 32, 0, stream>>>(xbf, wtv, bv, nullptr, vtb, BL_, D_, D_, 3);

  k_attn<<<B_ * H_ * (L_ / 16), 32, 0, stream>>>(qbf, kbf, vtb, erv, obf);

  dim3 gF1(BL_ / 32, DFC_ / 64);
  k_gemm<<<gF1, 32, 0, stream>>>(obf, w1t, b1, nullptr, hbf, BL_, DFC_, D_, 1);
  dim3 gF2(BL_ / 32, D_ / 64);
  k_gemm<<<gF2, 32, 0, stream>>>(hbf, w2t, b2, (float*)d_out, nullptr, BL_, D_, DFC_, 0);
}
